// TernaryLinear_31207232372816
// MI455X (gfx1250) — compile-verified, matrix operand order checked by reading, not run on hardware
//
#include <hip/hip_runtime.h>

typedef __attribute__((ext_vector_type(16))) _Float16 v16h;
typedef __attribute__((ext_vector_type(8)))  _Float16 v8h;
typedef __attribute__((ext_vector_type(4)))  _Float16 v4h;
typedef __attribute__((ext_vector_type(8)))  float    v8f;
typedef __attribute__((ext_vector_type(4)))  float    v4f;

#define IN_F   4096
#define OUT_F  16384
#define BATCH  8192
#define THRESH 0.3f

// Tiling: 128x128 block tile, K-step 32. 256 threads = 8 waves (wave32).
// Waves arranged 2(M) x 4(N); each wave owns 64x32 = 4x2 tiles of 16x16.
constexpr int BM = 128;
constexpr int BN = 128;
constexpr int BK = 32;
constexpr int LDSS = 40;              // halves per row: 32 + 8 pad (80B rows)
constexpr int ABUF = BM * LDSS;       // halves per A buffer
constexpr int BBUF = BN * LDSS;       // halves per B buffer

constexpr size_t X_ELEMS = (size_t)BATCH * IN_F;
constexpr size_t W_ELEMS = (size_t)OUT_F * IN_F;
constexpr size_t X_HALF_BYTES = X_ELEMS * 2;       // 64 MB
constexpr size_t W_HALF_BYTES = W_ELEMS * 2;       // 128 MB

// ---------------------------------------------------------------------------
// CDNA5 async global->LDS path (ASYNCcnt). Probe-confirmed signature:
//   __builtin_amdgcn_global_load_async_to_lds_b128(
//       AS1 int4* src, AS3 int4* dst, imm int offset, imm int cpol)
// ---------------------------------------------------------------------------
#if __has_builtin(__builtin_amdgcn_global_load_async_to_lds_b128) && \
    __has_builtin(__builtin_amdgcn_s_wait_asynccnt)
#define HAVE_ASYNC_LDS 1
typedef int vint4 __attribute__((vector_size(16)));
typedef __attribute__((address_space(1))) vint4 gv4i;
typedef __attribute__((address_space(3))) vint4 lv4i;
__device__ __forceinline__ void async_copy16(const void* g, void* l) {
    // Generic LDS pointer carries the LDS offset in its low 32 bits.
    __builtin_amdgcn_global_load_async_to_lds_b128(
        (gv4i*)(unsigned long long)g,
        (lv4i*)(unsigned int)(unsigned long long)l,
        0, 0);
}
#else
#define HAVE_ASYNC_LDS 0
#endif

// ---------------------------------------------------------------------------
// Pre-pass: fp32 -> f16 (optionally ternarized with fp32 threshold compare,
// bit-exact vs. the reference's dead-zone quantizer). 8 elements per thread.
// ---------------------------------------------------------------------------
template <bool TERN>
__global__ __launch_bounds__(256)
void convert_f16(const float* __restrict__ src, _Float16* __restrict__ dst)
{
    const size_t i = ((size_t)blockIdx.x * blockDim.x + threadIdx.x) * 8;
    v4f a = *(const v4f*)(src + i);
    v4f b = *(const v4f*)(src + i + 4);
    v8h h;
#pragma unroll
    for (int j = 0; j < 4; ++j) {
        float va = a[j], vb = b[j];
        if (TERN) {
            va = (va > THRESH) ? 1.0f : ((va < -THRESH) ? -1.0f : 0.0f);
            vb = (vb > THRESH) ? 1.0f : ((vb < -THRESH) ? -1.0f : 0.0f);
        }
        h[j]     = (_Float16)va;
        h[j + 4] = (_Float16)vb;
    }
    *(v8h*)(dst + i) = h;
}

// ---------------------------------------------------------------------------
// Shared compute step: 6 fragments (4 A + 2 B) from LDS, 8 WMMAs.
// Fragment layout per 16-bit A 16x32 spec:
//   lanes 0-15 : halves 0..7 = K0..7,  halves 8..15 = K16..23
//   lanes 16-31: halves 0..7 = K8..15, halves 8..15 = K24..31
// ---------------------------------------------------------------------------
__device__ __forceinline__ void wmma_step(const _Float16* Abuf, const _Float16* Bbuf,
                                          int wm, int wn, int lane, v8f acc[4][2])
{
    const int r   = lane & 15;
    const int kl0 = (lane < 16) ? 0 : 8;

    v16h afrag[4];
#pragma unroll
    for (int i = 0; i < 4; ++i) {
        const _Float16* p = Abuf + (wm * 64 + i * 16 + r) * LDSS;
        union { v16h v; v8h h[2]; } u;
        u.h[0] = *(const v8h*)(p + kl0);
        u.h[1] = *(const v8h*)(p + kl0 + 16);
        afrag[i] = u.v;
    }
    v16h bfrag[2];
#pragma unroll
    for (int j = 0; j < 2; ++j) {
        const _Float16* p = Bbuf + (wn * 32 + j * 16 + r) * LDSS;
        union { v16h v; v8h h[2]; } u;
        u.h[0] = *(const v8h*)(p + kl0);
        u.h[1] = *(const v8h*)(p + kl0 + 16);
        bfrag[j] = u.v;
    }

#pragma unroll
    for (int i = 0; i < 4; ++i)
#pragma unroll
        for (int j = 0; j < 2; ++j)
            acc[i][j] = __builtin_amdgcn_wmma_f32_16x16x32_f16(
                false, afrag[i], false, bfrag[j],
                (short)0, acc[i][j], false, false);
}

// ---------------------------------------------------------------------------
// Fast-path GEMM: pre-converted f16 operands, double-buffered LDS,
// async global->LDS copies when available (one barrier per K-step).
// ---------------------------------------------------------------------------
__global__ __launch_bounds__(256)
void ternary_gemm_f16(const _Float16* __restrict__ Xh,
                      const _Float16* __restrict__ Wh,
                      const float* __restrict__ bias,
                      float* __restrict__ Out)
{
    __shared__ _Float16 As[2 * ABUF];
    __shared__ _Float16 Bs[2 * BBUF];

    const int tid  = threadIdx.x;
    const int lane = tid & 31;
    const int wave = tid >> 5;
    const int wm   = wave & 1;      // 2 waves along M
    const int wn   = wave >> 1;     // 4 waves along N

    const int blockM = blockIdx.y * BM;
    const int blockN = blockIdx.x * BN;

    // Tile = 128 rows x 32 halves = 512 chunks of 8 halves (16B).
    // 256 threads -> 2 chunks each. c = tid + 256*i; row = c>>2; kc = (c&3)*8.
    const int row0 = tid >> 2,         kc0 = (tid & 3) * 8;
    const int row1 = (tid + 256) >> 2, kc1 = ((tid + 256) & 3) * 8;

    const _Float16* gA0 = Xh + (size_t)(blockM + row0) * IN_F + kc0;
    const _Float16* gA1 = Xh + (size_t)(blockM + row1) * IN_F + kc1;
    const _Float16* gB0 = Wh + (size_t)(blockN + row0) * IN_F + kc0;
    const _Float16* gB1 = Wh + (size_t)(blockN + row1) * IN_F + kc1;

    _Float16* lA0 = As + row0 * LDSS + kc0;
    _Float16* lA1 = As + row1 * LDSS + kc1;
    _Float16* lB0 = Bs + row0 * LDSS + kc0;
    _Float16* lB1 = Bs + row1 * LDSS + kc1;

    v8f acc[4][2] = {};

    // ---- Prologue: stage tile 0 into buffer 0 ----
#if HAVE_ASYNC_LDS
    async_copy16(gA0, lA0);
    async_copy16(gA1, lA1);
    async_copy16(gB0, lB0);
    async_copy16(gB1, lB1);
    __builtin_amdgcn_s_wait_asynccnt(0);
#else
    {
        v8h a0 = *(const v8h*)gA0, a1 = *(const v8h*)gA1;
        v8h b0 = *(const v8h*)gB0, b1 = *(const v8h*)gB1;
        *(v8h*)lA0 = a0; *(v8h*)lA1 = a1;
        *(v8h*)lB0 = b0; *(v8h*)lB1 = b1;
    }
#endif
    __syncthreads();

    const int nIter = IN_F / BK;   // 128
    for (int t = 0; t < nIter; ++t) {
        const int cur = t & 1;
        const int nxt = cur ^ 1;
        const int kNext = (t + 1) * BK;
        const bool hasNext = (kNext < IN_F);

#if HAVE_ASYNC_LDS
        if (hasNext) {
            // Bypass VGPRs: global -> LDS[nxt] on the ASYNCcnt path.
            async_copy16(gA0 + kNext, lA0 + nxt * ABUF);
            async_copy16(gA1 + kNext, lA1 + nxt * ABUF);
            async_copy16(gB0 + kNext, lB0 + nxt * BBUF);
            async_copy16(gB1 + kNext, lB1 + nxt * BBUF);
        }
#else
        v8h a0, a1, b0, b1;
        if (hasNext) {
            a0 = *(const v8h*)(gA0 + kNext);
            a1 = *(const v8h*)(gA1 + kNext);
            b0 = *(const v8h*)(gB0 + kNext);
            b1 = *(const v8h*)(gB1 + kNext);
        }
#endif

        // Compute on the current buffer while the next tile is in flight.
        wmma_step(As + cur * ABUF, Bs + cur * BBUF, wm, wn, lane, acc);

#if HAVE_ASYNC_LDS
        if (hasNext) __builtin_amdgcn_s_wait_asynccnt(0);
#else
        if (hasNext) {
            *(v8h*)(lA0 + nxt * ABUF) = a0;
            *(v8h*)(lA1 + nxt * ABUF) = a1;
            *(v8h*)(lB0 + nxt * BBUF) = b0;
            *(v8h*)(lB1 + nxt * BBUF) = b1;
        }
#endif
        __syncthreads();   // publish LDS[nxt]; reads of LDS[cur] are done
    }

    // Writeback. C/D layout: VGPR v, lane L -> M = v + 8*(L>=16), N = L&15.
    const int r   = lane & 15;
    const int hi8 = (lane >> 4) * 8;
#pragma unroll
    for (int j = 0; j < 2; ++j) {
        const int n    = blockN + wn * 32 + j * 16 + r;
        const float bv = bias[n];
#pragma unroll
        for (int i = 0; i < 4; ++i) {
            const int mbase = blockM + wm * 64 + i * 16 + hi8;
#pragma unroll
            for (int v = 0; v < 8; ++v) {
                Out[(size_t)(mbase + v) * OUT_F + n] = acc[i][j][v] + bv;
            }
        }
    }
}

// ---------------------------------------------------------------------------
// Fallback GEMM (no workspace): inline fp32->f16 conversion + ternarize,
// single-buffered LDS.
// ---------------------------------------------------------------------------
__global__ __launch_bounds__(256)
void ternary_gemm_inline(const float* __restrict__ X,
                         const float* __restrict__ W,
                         const float* __restrict__ bias,
                         float* __restrict__ Out)
{
    __shared__ _Float16 As[ABUF];
    __shared__ _Float16 Bs[BBUF];

    const int tid  = threadIdx.x;
    const int lane = tid & 31;
    const int wave = tid >> 5;
    const int wm   = wave & 1;
    const int wn   = wave >> 1;

    const int blockM = blockIdx.y * BM;
    const int blockN = blockIdx.x * BN;

    v8f acc[4][2] = {};

    for (int k0 = 0; k0 < IN_F; k0 += BK) {
        v4f areg[4], breg[4];
#pragma unroll
        for (int i = 0; i < 4; ++i) {
            const int c   = tid + 256 * i;
            const int row = c >> 3;
            const int kc  = (c & 7) * 4;
            areg[i] = *(const v4f*)(X + (size_t)(blockM + row) * IN_F + k0 + kc);
            breg[i] = *(const v4f*)(W + (size_t)(blockN + row) * IN_F + k0 + kc);
        }
        __syncthreads();
#pragma unroll
        for (int i = 0; i < 4; ++i) {
            const int c   = tid + 256 * i;
            const int row = c >> 3;
            const int kc  = (c & 7) * 4;
            v4h ah, bh;
#pragma unroll
            for (int j = 0; j < 4; ++j) {
                ah[j] = (_Float16)areg[i][j];
                float w = breg[i][j];
                float q = (w > THRESH) ? 1.0f : ((w < -THRESH) ? -1.0f : 0.0f);
                bh[j] = (_Float16)q;
            }
            *(v4h*)(As + row * LDSS + kc) = ah;
            *(v4h*)(Bs + row * LDSS + kc) = bh;
        }
        __syncthreads();

        wmma_step(As, Bs, wm, wn, lane, acc);
    }

    const int r   = lane & 15;
    const int hi8 = (lane >> 4) * 8;
#pragma unroll
    for (int j = 0; j < 2; ++j) {
        const int n    = blockN + wn * 32 + j * 16 + r;
        const float bv = bias[n];
#pragma unroll
        for (int i = 0; i < 4; ++i) {
            const int mbase = blockM + wm * 64 + i * 16 + hi8;
#pragma unroll
            for (int v = 0; v < 8; ++v) {
                Out[(size_t)(mbase + v) * OUT_F + n] = acc[i][j][v] + bv;
            }
        }
    }
}

extern "C" void kernel_launch(void* const* d_in, const int* in_sizes, int n_in,
                              void* d_out, int out_size, void* d_ws, size_t ws_size,
                              hipStream_t stream) {
    const float* x    = (const float*)d_in[0];   // [8192, 4096]
    const float* w    = (const float*)d_in[1];   // [16384, 4096]
    const float* bias = (const float*)d_in[2];   // [16384]
    float* out        = (float*)d_out;           // [8192, 16384]

    dim3 grid(OUT_F / BN, BATCH / BM);           // (128, 64)
    dim3 block(256);

    if (ws_size >= X_HALF_BYTES + W_HALF_BYTES && d_ws != nullptr) {
        _Float16* xh = (_Float16*)d_ws;
        _Float16* wh = (_Float16*)((char*)d_ws + X_HALF_BYTES);
        convert_f16<false><<<dim3((unsigned)(X_ELEMS / 8 / 256)), block, 0, stream>>>(x, xh);
        convert_f16<true ><<<dim3((unsigned)(W_ELEMS / 8 / 256)), block, 0, stream>>>(w, wh);
        ternary_gemm_f16<<<grid, block, 0, stream>>>(xh, wh, bias, out);
    } else {
        ternary_gemm_inline<<<grid, block, 0, stream>>>(x, w, bias, out);
    }
}
